// VectorQuantizer_83580063580937
// MI455X (gfx1250) — matrix operand order repaired and checked by reading, not compiled, not run on hardware
//
#include <hip/hip_runtime.h>
#include <hip/hip_bf16.h>

typedef __attribute__((ext_vector_type(16))) _Float16     v16h;
typedef __attribute__((ext_vector_type(8)))  float        v8f;
typedef __attribute__((ext_vector_type(4)))  float        f32x4;
typedef __attribute__((ext_vector_type(4)))  unsigned int u32x4;
typedef __attribute__((ext_vector_type(8)))  unsigned int u32x8;
typedef __attribute__((ext_vector_type(4)))  int          i32x4;
typedef __attribute__((ext_vector_type(8)))  int          i32x8;

#define NUM_K   512
#define DIM     64
#define BATCH   32
#define TLEN    4096
#define NROWS   (BATCH * TLEN)        // 131072 flat vectors
#define QELEMS  (BATCH * DIM * TLEN)  // 8388608 quantized outputs
#define LOSS_OFF QELEMS
#define IDX_OFF  (QELEMS + 1)

#define WAVES_PER_BLOCK 8
#define TILES_PER_WAVE  2
#define BLOCK_T  (WAVES_PER_BLOCK * TILES_PER_WAVE * 16)  // 256 t per block
#define NBLOCKS  (NROWS / BLOCK_T)                        // 512

#if defined(__has_builtin)
#if __has_builtin(__builtin_amdgcn_tensor_load_to_lds) && \
    __has_builtin(__builtin_amdgcn_s_wait_tensorcnt)
#define HAVE_TDM 1
#endif
#endif

union FragU { v16h h; u32x4 u[2]; };

// score = (2.0 + 0.5||e||^2) - x.e  computed as WMMA D = A*B + C with A = -x.
// All scores land in ~(1.7, 2.3) > 0, so float bits order as unsigned ints and
// the low 9 mantissa bits can carry the code index for branchless argmin.
__global__ __launch_bounds__(256) void vq_main_kernel(
    const float* __restrict__ x,     // [B, D, T] fp32
    const float* __restrict__ cbg,   // [K, D] fp32
    float* __restrict__ dout,        // [QELEMS quantized][1 loss][NROWS idx]
    float* __restrict__ lossAcc)     // d_ws accumulator
{
    __shared__ __align__(16) float    xtile[DIM * BLOCK_T];  // 64KB fp32 x tile
    __shared__ __align__(16) _Float16 cb[NUM_K * DIM];       // 64KB f16 codebook
    __shared__ float bias[NUM_K];                            // 2 + 0.5*||e||^2
    __shared__ int   idxbuf[WAVES_PER_BLOCK * 16];

    const int tid  = threadIdx.x;
    const int lane = tid & 31;
    const int w    = tid >> 5;
    const int hb   = lane >> 4;      // half-wave selector (ISA lane layout)
    const int col  = lane & 15;      // column within 16x16 tile

    const int blockT0 = blockIdx.x * BLOCK_T;
    const int b       = blockT0 / TLEN;          // block stays inside one batch
    const int t0blk   = blockT0 - b * TLEN;
    const float* xg   = x + (size_t)b * DIM * TLEN + t0blk;  // tile start

#if HAVE_TDM
    // ---- TDM: async DMA of [64 d x 256 t] fp32 tile into LDS (wave 0) ----
    if (w == 0) {
        const unsigned long long ga = (unsigned long long)(uintptr_t)xg;
        const unsigned ldsoff = (unsigned)(uintptr_t)xtile;   // addr[31:0] = LDS offset
        u32x4 g0;
        g0.x = 1u;                                     // count=1, user mode
        g0.y = ldsoff;                                 // lds_addr
        g0.z = (unsigned)(ga & 0xFFFFFFFFu);           // global_addr[31:0]
        g0.w = (unsigned)((ga >> 32) & 0x01FFFFFFu) | 0x80000000u;  // [56:32]|type=2
        i32x8 g1;
        g1[0] = 0x20000;                   // data_size=2 (4B), no mask/pad/iter
        g1[1] = (int)(4096u << 16);        // tensor_dim0 = 4096 (t extent)
        g1[2] = (int)(64u << 16);          // tensor_dim1 = 64 (d extent)
        g1[3] = (int)((unsigned)BLOCK_T << 16);  // tile_dim0 = 256
        g1[4] = 64;                        // tile_dim1 = 64, tile_dim2 = 0
        g1[5] = 4096;                      // tensor_dim0_stride = T
        g1[6] = 0;
        g1[7] = 0;
        i32x4 z4 = {0, 0, 0, 0};
#if __clang_major__ >= 23
        i32x8 z8 = {0, 0, 0, 0, 0, 0, 0, 0};
        __builtin_amdgcn_tensor_load_to_lds(g0, g1, z4, z4, z8, 0);
#else
        __builtin_amdgcn_tensor_load_to_lds(g0, g1, z4, z4, 0);
#endif
    }
#else
    for (int i = tid; i < DIM * BLOCK_T; i += 256) {
        const int d = i >> 8;               // / BLOCK_T
        const int t = i & (BLOCK_T - 1);
        xtile[i] = xg[(size_t)d * TLEN + t];
    }
#endif

    // ---- cooperative codebook convert fp32 -> f16 into LDS (overlaps TDM) ----
    for (int i = tid * 4; i < NUM_K * DIM; i += 256 * 4) {
        f32x4 v = *(const f32x4*)(cbg + i);
        cb[i + 0] = (_Float16)v.x;
        cb[i + 1] = (_Float16)v.y;
        cb[i + 2] = (_Float16)v.z;
        cb[i + 3] = (_Float16)v.w;
    }
    // ---- exact fp32 bias: 2.0 + 0.5*||e_k||^2 (keeps scores positive) ----
    for (int k = tid; k < NUM_K; k += 256) {
        const float* row = cbg + k * DIM;
        float s = 0.f;
        #pragma unroll 8
        for (int d = 0; d < DIM; ++d) { float e = row[d]; s += e * e; }
        bias[k] = 2.0f + 0.5f * s;
    }
#if HAVE_TDM
    if (w == 0) __builtin_amdgcn_s_wait_tensorcnt(0);
#endif
    __syncthreads();

    const u32x4* cbv = (const u32x4*)cb;     // 16B granules of f16 codebook
    float* outb   = dout + (size_t)b * DIM * TLEN;
    float* idxOut = dout + IDX_OFF;
    float  lsum   = 0.f;

    for (int r = 0; r < TILES_PER_WAVE; ++r) {
        const int tl0 = (w * TILES_PER_WAVE + r) * 16;  // local t of row tile
        const int n0  = blockT0 + tl0;                  // global flat row base
        const int tl  = tl0 + col;

        // ---- A fragments from LDS: -x in f16, ISA 16x32 interleaved layout ----
        FragU A0, A1;
        #pragma unroll
        for (int e = 0; e < 16; ++e) {
            const int d = (e < 8) ? (hb * 8 + e) : (16 + hb * 8 + (e - 8));
            A0.h[e] = (_Float16)(-xtile[d        * BLOCK_T + tl]);
            A1.h[e] = (_Float16)(-xtile[(d + 32) * BLOCK_T + tl]);
        }

        u32x8 mink;
        #pragma unroll
        for (int j = 0; j < 8; ++j) mink[j] = 0xFFFFFFFFu;

        // ---- sweep all 32 codebook tiles ----
        // Bounded unroll: 2 independent acc/mink chains hide the WMMA->VALU
        // hazard slots without blowing the register file (full unroll spilled).
        #pragma unroll 2
        for (int c0 = 0; c0 < NUM_K; c0 += 16) {
            const int c = c0 + col;          // this lane's code column
            // B 32x16 f16: lane n holds K=0..7,16..23, lane n+16 holds K=8..15,24..31
            FragU B0, B1;
            B0.u[0] = cbv[c * 8 + hb];
            B0.u[1] = cbv[c * 8 + 2 + hb];
            B1.u[0] = cbv[c * 8 + 4 + hb];
            B1.u[1] = cbv[c * 8 + 6 + hb];

            const float bc = bias[c];
            v8f acc;
            #pragma unroll
            for (int j = 0; j < 8; ++j) acc[j] = bc;  // C = 2 + 0.5||e||^2

            acc = __builtin_amdgcn_wmma_f32_16x16x32_f16(
                false, A0.h, false, B0.h, (short)0, acc, false, false);
            acc = __builtin_amdgcn_wmma_f32_16x16x32_f16(
                false, A1.h, false, B1.h, (short)0, acc, false, false);

            // branchless argmin, whole-vector form: pack this lane's code index
            // into the 9 low mantissa bits of each of the 8 row scores, then
            // elementwise unsigned min against the running keys.
            u32x8 keys = (__builtin_bit_cast(u32x8, acc) & 0xFFFFFE00u)
                         | (unsigned)c;
            mink = __builtin_elementwise_min(mink, keys);
        }

        // ---- branchless reduce across the 16 lanes of each half-wave ----
        #pragma unroll
        for (int j = 0; j < 8; ++j) {
            unsigned mj = mink[j];
            #pragma unroll
            for (int off = 1; off < 16; off <<= 1) {
                unsigned ok = (unsigned)__shfl_xor((int)mj, off, 16);
                mj = ok < mj ? ok : mj;
            }
            mink[j] = mj;
        }

        // lanes 0-15 hold rows 0-7 (slot j = row); lanes 16-31 rows 8-15
        if (lane == 0 || lane == 16) {
            #pragma unroll
            for (int j = 0; j < 8; ++j) {
                const int row = hb * 8 + j;
                const int idx = (int)(mink[j] & 0x1FFu);
                idxbuf[w * 16 + row] = idx;
                idxOut[n0 + row] = (float)idx;
            }
        }

        // ---- output + loss pass (full fp32, coalesced [d][t] stores) ----
        const int myIdx = idxbuf[w * 16 + col];   // code for row t0+col
        #pragma unroll 4
        for (int it = 0; it < 32; ++it) {
            const int d  = it * 2 + hb;
            const float q  = cbg[myIdx * DIM + d];
            const float xv = xtile[d * BLOCK_T + tl];
            const float dq = q - xv;
            outb[(size_t)d * TLEN + t0blk + tl] = xv + dq;  // x + (q - x)
            lsum += dq * dq;
        }
    }

    // ---- wave-reduce squared error, one atomic per wave ----
    #pragma unroll
    for (int off = 1; off < 32; off <<= 1) lsum += __shfl_xor(lsum, off, 32);
    if (lane == 0) atomicAdd(lossAcc, lsum);
}

__global__ void vq_zero_kernel(float* lossAcc) {
    if (threadIdx.x == 0) lossAcc[0] = 0.f;
}

__global__ void vq_final_kernel(const float* lossAcc, float* dout) {
    // loss = q_latent + 0.25*e_latent = 1.25 * mean((q-x)^2)
    if (threadIdx.x == 0)
        dout[LOSS_OFF] = lossAcc[0] * (1.25f / (float)QELEMS);
}

extern "C" void kernel_launch(void* const* d_in, const int* in_sizes, int n_in,
                              void* d_out, int out_size, void* d_ws, size_t ws_size,
                              hipStream_t stream) {
    const float* x   = (const float*)d_in[0];   // [B, D, T] fp32
    const float* cbg = (const float*)d_in[1];   // [K, D] fp32
    float* dout = (float*)d_out;
    float* acc  = (float*)d_ws;

    vq_zero_kernel<<<1, 64, 0, stream>>>(acc);
    vq_main_kernel<<<NBLOCKS, 256, 0, stream>>>(x, cbg, dout, acc);
    vq_final_kernel<<<1, 64, 0, stream>>>(acc, dout);
}
